// RelativeMultiHeadAttention_1864015807176
// MI455X (gfx1250) — matrix operand order repaired and checked by reading, not compiled, not run on hardware
//
#include <hip/hip_runtime.h>
#include <hip/hip_bf16.h>

// ---------------- types & helpers ----------------
typedef __attribute__((ext_vector_type(16))) __bf16        v16bf;
typedef __attribute__((ext_vector_type(8)))  float         v8f;
typedef __attribute__((ext_vector_type(4)))  unsigned int  v4u;
typedef __attribute__((ext_vector_type(8)))  unsigned int  v8u;

#define WB  4
#define WS  1024
#define WD  512
#define WH  16
#define WDH 32

__device__ __forceinline__ unsigned short f2bf(float f) {
  unsigned int u = __builtin_bit_cast(unsigned int, f);
  u += 0x7FFFu + ((u >> 16) & 1u);                 // round-to-nearest-even
  return (unsigned short)(u >> 16);
}
__device__ __forceinline__ __bf16 f2bfh(float f) {
  unsigned short s = f2bf(f);
  return __builtin_bit_cast(__bf16, s);
}

__device__ __forceinline__ v8f wmma_bf16(v16bf a, v16bf b, v8f c) {
  // D = A(16x32 bf16) x B(32x16 bf16) + C(16x16 f32)
  return __builtin_amdgcn_wmma_f32_16x16x32_bf16(false, a, false, b,
                                                 (short)0, c, false, false);
}

// ---------------- kernel 1: fused QKVP projections ----------------
// z=0: q -> qu=(q@Wq+bq)+u_bias, qv=(q@Wq+bq)+v_bias
// z=1: k -> kb ; z=2: v -> vb ; z=3: pos -> pb (no bias)
__global__ __launch_bounds__(256)
void proj_kernel(const float* __restrict__ query, const float* __restrict__ key,
                 const float* __restrict__ value, const float* __restrict__ pose,
                 const float* __restrict__ Wq, const float* __restrict__ Wk,
                 const float* __restrict__ Wv, const float* __restrict__ Wp,
                 const float* __restrict__ bq, const float* __restrict__ bk,
                 const float* __restrict__ bv,
                 const float* __restrict__ u_bias, const float* __restrict__ v_bias,
                 unsigned short* __restrict__ qu, unsigned short* __restrict__ qv,
                 unsigned short* __restrict__ kb, unsigned short* __restrict__ vb,
                 unsigned short* __restrict__ pb)
{
  __shared__ unsigned short lds_a[64 * 32];
  __shared__ unsigned short lds_w[32 * 32];

  const int z  = blockIdx.z;
  const float* Ap = (z == 0) ? query : (z == 1) ? key : (z == 2) ? value : pose;
  const float* Wm = (z == 0) ? Wq    : (z == 1) ? Wk  : (z == 2) ? Wv    : Wp;

  const int m0   = blockIdx.x * 64;
  const int n0   = blockIdx.y * 32;
  const int tid  = threadIdx.x;
  const int wave = tid >> 5;
  const int lane = tid & 31;
  const int wm   = (wave & 3) * 16;   // wave row within block tile
  const int wn   = (wave >> 2) * 16;  // wave col within block tile
  const int lrow = lane & 15;
  const int hi   = lane >> 4;
  const int half8 = hi * 8;
  const int kb16  = hi * 16;

  v8f acc = {};
  for (int kk = 0; kk < WD; kk += 32) {
    __syncthreads();
    for (int idx = tid; idx < 64 * 32; idx += 256) {
      int r = idx >> 5, c = idx & 31;
      lds_a[idx] = f2bf(Ap[(size_t)(m0 + r) * WD + kk + c]);
    }
    for (int idx = tid; idx < 32 * 32; idx += 256) {
      int r = idx >> 5, c = idx & 31;
      lds_w[idx] = f2bf(Wm[(size_t)(kk + r) * WD + n0 + c]);
    }
    __syncthreads();

    v16bf af, bf;
    {
      const int row = wm + lrow;
      #pragma unroll
      for (int i = 0; i < 4; i++) {
        int k0 = kb16 + 2 * i;
        af[2*i]       = __builtin_bit_cast(__bf16, lds_a[row * 32 + k0]);
        af[2*i+1]     = __builtin_bit_cast(__bf16, lds_a[row * 32 + k0 + 1]);
        int k1 = kb16 + 8 + 2 * i;
        af[2*(i+4)]   = __builtin_bit_cast(__bf16, lds_a[row * 32 + k1]);
        af[2*(i+4)+1] = __builtin_bit_cast(__bf16, lds_a[row * 32 + k1 + 1]);
      }
      const int col = wn + lrow;
      #pragma unroll
      for (int i = 0; i < 8; i++) {
        bf[2*i]   = __builtin_bit_cast(__bf16, lds_w[(kb16 + 2*i) * 32 + col]);
        bf[2*i+1] = __builtin_bit_cast(__bf16, lds_w[(kb16 + 2*i + 1) * 32 + col]);
      }
    }
    acc = wmma_bf16(af, bf, acc);
  }

  #pragma unroll
  for (int v = 0; v < 8; v++) {
    const int m  = m0 + wm + v + half8;
    const int n  = n0 + wn + lrow;
    const int bI = m >> 10, s = m & 1023;
    const int h  = n >> 5,  dh = n & 31;
    const size_t o = (((size_t)(bI * WH + h) * WS) + s) * WDH + dh;
    float val = acc[v];
    if (z == 0) {
      float q0 = val + bq[n];
      qu[o] = f2bf(q0 + u_bias[h * WDH + dh]);
      qv[o] = f2bf(q0 + v_bias[h * WDH + dh]);
    } else if (z == 1) {
      kb[o] = f2bf(val + bk[n]);
    } else if (z == 2) {
      vb[o] = f2bf(val + bv[n]);
    } else {
      pb[o] = f2bf(val);
    }
  }
}

// ---------------- kernel 2: fused rel-shift flash attention ----------------
// one wave per (b, h, 16-row query block); streams over t, never materializes S x T
__global__ __launch_bounds__(32)
void attn_kernel(const unsigned short* __restrict__ qu,
                 const unsigned short* __restrict__ qv,
                 const unsigned short* __restrict__ kb,
                 const unsigned short* __restrict__ vb,
                 const unsigned short* __restrict__ pb,
                 float* __restrict__ ctx)
{
  __shared__ float slabA[16 * 32];
  __shared__ float slabB[16 * 32];
  __shared__ float pt[16 * 32];

  const int i0   = blockIdx.x * 16;
  const int h    = blockIdx.y;
  const int b    = blockIdx.z;
  const int lane = threadIdx.x & 31;
  const int lrow = lane & 15;
  const int hi   = lane >> 4;
  const int half8 = hi * 8;
  const int kb16  = hi * 16;
  const float scale = 0.044194173824159216f;  // 1/sqrt(512)

  const size_t headoff = (size_t)(b * WH + h) * WS * WDH;
  const unsigned short* quh = qu + headoff;
  const unsigned short* qvh = qv + headoff;
  const unsigned short* kbh = kb + headoff;
  const unsigned short* vbh = vb + headoff;
  const unsigned short* pbh = pb + headoff;

  // A-fragment loader: row per lane, dh chunk [kb16, kb16+16)
  auto loadA = [&](const unsigned short* base, int row, bool guard) -> v16bf {
    v16bf a;
    if (guard && (row < 0 || row >= WS)) {
      #pragma unroll
      for (int e = 0; e < 16; e++) a[e] = __builtin_bit_cast(__bf16, (unsigned short)0);
      return a;
    }
    const unsigned short* p = base + (size_t)row * WDH;
    v4u lo  = *(const v4u*)(p + kb16);
    v4u hi4 = *(const v4u*)(p + kb16 + 8);
    v8u w;
    #pragma unroll
    for (int i = 0; i < 4; i++) { w[i] = lo[i]; w[i + 4] = hi4[i]; }
    return __builtin_bit_cast(v16bf, w);
  };
  // B-fragment loader where K-dim is dh: lane covers col = t row, dims kb16..kb16+15
  auto loadB = [&](const unsigned short* base, int trow) -> v16bf {
    v16bf bfv;
    if (trow < 0 || trow >= WS) {
      #pragma unroll
      for (int e = 0; e < 16; e++) bfv[e] = __builtin_bit_cast(__bf16, (unsigned short)0);
      return bfv;
    }
    const v8u* p = (const v8u*)(base + (size_t)trow * WDH + kb16);
    return __builtin_bit_cast(v16bf, *p);
  };
  // B-fragment loader for V: K-dim is t, col = dh
  auto loadV = [&](int jb, int dhbase) -> v16bf {
    v16bf vf;
    const int dh = dhbase + lrow;
    #pragma unroll
    for (int i = 0; i < 8; i++) {
      int t = jb + kb16 + 2 * i;
      vf[2*i]   = __builtin_bit_cast(__bf16, vbh[(size_t)t * WDH + dh]);
      vf[2*i+1] = __builtin_bit_cast(__bf16, vbh[(size_t)(t + 1) * WDH + dh]);
    }
    return vf;
  };

  const v16bf Au  = loadA(quh, i0 + lrow, false);
  const v16bf Av  = loadA(qvh, i0 + lrow, false);
  const v16bf Av2 = loadA(qvh, i0 + 1 + lrow, true);  // row i+1 for wrap branch

  v8f O0 = {}, O1 = {};
  float mrow[8], lrowsum[8];
  #pragma unroll
  for (int v = 0; v < 8; v++) { mrow[v] = -1e30f; lrowsum[v] = 0.f; }

  for (int jb = 0; jb < WS; jb += 32) {
    v8f L[2];
    #pragma unroll
    for (int c16 = 0; c16 < 2; c16++) {
      const int j0 = jb + c16 * 16;
      // ----- content score tile (one WMMA, K = DH = 32) -----
      v8f scr = {};
      scr = wmma_bf16(Au, loadB(kbh, j0 + lrow), scr);

      // ----- shifted position score via PS slabs -----
      const int d = j0 - i0;
      const bool below = (d <= -15);  // whole tile j<=i
      const bool above = (d >= 17);   // whole tile j>=i+2
      const int tbA = WS - 16 - i0 + j0;  // slab A window base (rows i0..i0+15)
      const int tbB = j0 - i0 - 17;       // slab B window base (rows i0+1..i0+16)
      if (!above) {
        v8f p0 = {}, p1 = {};
        p0 = wmma_bf16(Av, loadB(pbh, tbA + lrow), p0);
        p1 = wmma_bf16(Av, loadB(pbh, tbA + 16 + lrow), p1);
        #pragma unroll
        for (int v = 0; v < 8; v++) {
          slabA[(v + half8) * 32 + lrow]      = p0[v];
          slabA[(v + half8) * 32 + 16 + lrow] = p1[v];
        }
      }
      if (!below) {
        v8f p0 = {}, p1 = {};
        p0 = wmma_bf16(Av2, loadB(pbh, tbB + lrow), p0);
        p1 = wmma_bf16(Av2, loadB(pbh, tbB + 16 + lrow), p1);
        #pragma unroll
        for (int v = 0; v < 8; v++) {
          slabB[(v + half8) * 32 + lrow]      = p0[v];
          slabB[(v + half8) * 32 + 16 + lrow] = p1[v];
        }
      }
      __syncthreads();
      // diagonal gather: slab column = c - r + 15 for both branches
      #pragma unroll
      for (int v = 0; v < 8; v++) {
        const int r = v + half8;
        const int scol = lrow - r + 15;
        float pv;
        if (below)      pv = slabA[r * 32 + scol];
        else if (above) pv = slabB[r * 32 + scol];
        else {
          const int jg = j0 + lrow, ig = i0 + r;
          pv = (jg <= ig)     ? slabA[r * 32 + scol]
             : (jg == ig + 1) ? 0.f
                              : slabB[r * 32 + scol];
        }
        L[c16][v] = (scr[v] + pv) * scale;
      }
      __syncthreads();
    }

    // ----- flash softmax update over 32 columns -----
    float mn[8];
    #pragma unroll
    for (int v = 0; v < 8; v++) {
      float cand = fmaxf(L[0][v], L[1][v]);
      #pragma unroll
      for (int s = 1; s < 16; s <<= 1)
        cand = fmaxf(cand, __shfl_xor(cand, s, 32));
      mn[v] = fmaxf(mrow[v], cand);
    }
    v8f P0, P1;
    float rsum[8];
    #pragma unroll
    for (int v = 0; v < 8; v++) {
      P0[v] = __expf(L[0][v] - mn[v]);
      P1[v] = __expf(L[1][v] - mn[v]);
      float s = P0[v] + P1[v];
      #pragma unroll
      for (int sh = 1; sh < 16; sh <<= 1)
        s += __shfl_xor(s, sh, 32);
      rsum[v] = s;
    }
    #pragma unroll
    for (int v = 0; v < 8; v++) {
      float corr = __expf(mrow[v] - mn[v]);
      lrowsum[v] = lrowsum[v] * corr + rsum[v];
      mrow[v] = mn[v];
      O0[v] *= corr;
      O1[v] *= corr;
      pt[(v + half8) * 32 + lrow]      = P0[v];
      pt[(v + half8) * 32 + 16 + lrow] = P1[v];
    }
    __syncthreads();
    // C/D layout -> A layout transpose of P through LDS
    v16bf pA;
    #pragma unroll
    for (int i = 0; i < 4; i++) {
      int k0 = kb16 + 2 * i;
      pA[2*i]       = f2bfh(pt[lrow * 32 + k0]);
      pA[2*i+1]     = f2bfh(pt[lrow * 32 + k0 + 1]);
      int k1 = kb16 + 8 + 2 * i;
      pA[2*(i+4)]   = f2bfh(pt[lrow * 32 + k1]);
      pA[2*(i+4)+1] = f2bfh(pt[lrow * 32 + k1 + 1]);
    }
    __syncthreads();
    // ----- O += P~ x V (K = 32 t-values) -----
    O0 = wmma_bf16(pA, loadV(jb, 0),  O0);
    O1 = wmma_bf16(pA, loadV(jb, 16), O1);
  }

  // finalize: divide by row sums, write context [B,S,D] (D index = h*32+dh)
  #pragma unroll
  for (int v = 0; v < 8; v++) {
    const int r = v + half8;
    const float inv = 1.0f / lrowsum[v];
    const int s = i0 + r;
    const size_t o = (((size_t)b * WS) + s) * WD + h * WDH;
    ctx[o + lrow]      = O0[v] * inv;
    ctx[o + 16 + lrow] = O1[v] * inv;
  }
}

// ---------------- kernel 3: output projection ----------------
__global__ __launch_bounds__(256)
void outproj_kernel(const float* __restrict__ ctx, const float* __restrict__ Wo,
                    const float* __restrict__ bo, float* __restrict__ out)
{
  __shared__ unsigned short lds_a[64 * 32];
  __shared__ unsigned short lds_w[32 * 32];

  const int m0   = blockIdx.x * 64;
  const int n0   = blockIdx.y * 32;
  const int tid  = threadIdx.x;
  const int wave = tid >> 5;
  const int lane = tid & 31;
  const int wm   = (wave & 3) * 16;
  const int wn   = (wave >> 2) * 16;
  const int lrow = lane & 15;
  const int hi   = lane >> 4;
  const int half8 = hi * 8;
  const int kb16  = hi * 16;

  v8f acc = {};
  for (int kk = 0; kk < WD; kk += 32) {
    __syncthreads();
    for (int idx = tid; idx < 64 * 32; idx += 256) {
      int r = idx >> 5, c = idx & 31;
      lds_a[idx] = f2bf(ctx[(size_t)(m0 + r) * WD + kk + c]);
    }
    for (int idx = tid; idx < 32 * 32; idx += 256) {
      int r = idx >> 5, c = idx & 31;
      lds_w[idx] = f2bf(Wo[(size_t)(kk + r) * WD + n0 + c]);
    }
    __syncthreads();

    v16bf af, bf;
    {
      const int row = wm + lrow;
      #pragma unroll
      for (int i = 0; i < 4; i++) {
        int k0 = kb16 + 2 * i;
        af[2*i]       = __builtin_bit_cast(__bf16, lds_a[row * 32 + k0]);
        af[2*i+1]     = __builtin_bit_cast(__bf16, lds_a[row * 32 + k0 + 1]);
        int k1 = kb16 + 8 + 2 * i;
        af[2*(i+4)]   = __builtin_bit_cast(__bf16, lds_a[row * 32 + k1]);
        af[2*(i+4)+1] = __builtin_bit_cast(__bf16, lds_a[row * 32 + k1 + 1]);
      }
      const int col = wn + lrow;
      #pragma unroll
      for (int i = 0; i < 8; i++) {
        bf[2*i]   = __builtin_bit_cast(__bf16, lds_w[(kb16 + 2*i) * 32 + col]);
        bf[2*i+1] = __builtin_bit_cast(__bf16, lds_w[(kb16 + 2*i + 1) * 32 + col]);
      }
    }
    acc = wmma_bf16(af, bf, acc);
  }

  #pragma unroll
  for (int v = 0; v < 8; v++) {
    const int m = m0 + wm + v + half8;
    const int n = n0 + wn + lrow;
    out[(size_t)m * WD + n] = acc[v] + bo[n];
  }
}

// ---------------- launch ----------------
extern "C" void kernel_launch(void* const* d_in, const int* in_sizes, int n_in,
                              void* d_out, int out_size, void* d_ws, size_t ws_size,
                              hipStream_t stream) {
  (void)in_sizes; (void)n_in; (void)out_size; (void)ws_size;
  const float* query = (const float*)d_in[0];
  const float* key   = (const float*)d_in[1];
  const float* value = (const float*)d_in[2];
  const float* pose  = (const float*)d_in[3];
  const float* Wq    = (const float*)d_in[4];
  const float* bq    = (const float*)d_in[5];
  const float* Wk    = (const float*)d_in[6];
  const float* bk    = (const float*)d_in[7];
  const float* Wv    = (const float*)d_in[8];
  const float* bv    = (const float*)d_in[9];
  const float* Wp    = (const float*)d_in[10];
  const float* ub    = (const float*)d_in[11];
  const float* vbias = (const float*)d_in[12];
  const float* Wo    = (const float*)d_in[13];
  const float* bo    = (const float*)d_in[14];
  float* out = (float*)d_out;

  // workspace layout (bf16 tensors in [B,H,S,DH], ctx fp32 in [B,S,D]) = 28 MB
  const size_t nEl = (size_t)WB * WH * WS * WDH;  // 2M elements
  unsigned short* qu = (unsigned short*)d_ws;
  unsigned short* qv = qu + nEl;
  unsigned short* kb = qv + nEl;
  unsigned short* pb = kb + nEl;
  unsigned short* vb = pb + nEl;
  float* ctx = (float*)(vb + nEl);

  dim3 g1(WB * WS / 64, WD / 32, 4);
  proj_kernel<<<g1, 256, 0, stream>>>(query, key, value, pose, Wq, Wk, Wv, Wp,
                                      bq, bk, bv, ub, vbias,
                                      qu, qv, kb, vb, pb);

  dim3 g2(WS / 16, WH, WB);
  attn_kernel<<<g2, 32, 0, stream>>>(qu, qv, kb, vb, pb, ctx);

  dim3 g3(WB * WS / 64, WD / 32, 1);
  outproj_kernel<<<g3, 256, 0, stream>>>(ctx, Wo, bo, out);
}
